// TransformerBlock_21827023798579
// MI455X (gfx1250) — matrix operand order repaired and checked by reading, not compiled, not run on hardware
//
#include <hip/hip_runtime.h>
#include <math.h>
#include <stdint.h>

// ---------------- problem constants ----------------
#define BB   4
#define NN   1024
#define DD   1024
#define HH   16
#define HDD  64
#define MLPD 4096
#define ROWS (BB*NN)   // 4096

typedef __bf16 bf16;
typedef __attribute__((ext_vector_type(8)))  bf16  v8bf;
typedef __attribute__((ext_vector_type(16))) bf16  v16bf;
typedef __attribute__((ext_vector_type(8)))  float v8f;

// ---------------- CDNA5 async copy: global -> LDS (ASYNCcnt) --------------
__device__ inline void async_copy16(const bf16* gptr, bf16* lptr) {
  unsigned lds = (unsigned)(uintptr_t)lptr;              // flat LDS addr low 32 bits
  unsigned long long ga = (unsigned long long)(uintptr_t)gptr;
  asm volatile("global_load_async_to_lds_b128 %0, %1, off"
               :: "v"(lds), "v"(ga) : "memory");
}
__device__ inline void wait_async0() {
  asm volatile("s_wait_asynccnt 0x0" ::: "memory");
}

// ---------------- wave32 xor-reduce within 16-lane halves (ds_swizzle) ----
#define XS(v, m) __int_as_float(__builtin_amdgcn_ds_swizzle(__float_as_int(v), (((m) << 10) | 0x1f)))

__device__ inline float red16_max(float v) {
  v = fmaxf(v, XS(v, 1));
  v = fmaxf(v, XS(v, 2));
  v = fmaxf(v, XS(v, 4));
  v = fmaxf(v, XS(v, 8));
  return v;
}
__device__ inline float red16_sum(float v) {
  v = v + XS(v, 1);
  v = v + XS(v, 2);
  v = v + XS(v, 4);
  v = v + XS(v, 8);
  return v;
}

// ---------------- WMMA helpers -------------------------------------------
__device__ inline v8f wmma_bf16(v16bf a, v16bf b, v8f c) {
  return __builtin_amdgcn_wmma_f32_16x16x32_bf16(false, a, false, b, (short)0, c, false, false);
}

// A-fragment (16x32, row-major source, K contiguous):
// lane l: row m0 + (l&15); halves at kk+off+{0..7} and kk+off+16+{0..7}, off = 8*(l>=16)
__device__ inline v16bf load_frag_A(const bf16* base, size_t stride, int m0, int kk, int lane) {
  int m = m0 + (lane & 15);
  int off = (lane < 16) ? 0 : 8;
  const bf16* p = base + (size_t)m * stride + (size_t)(kk + off);
  v8bf lo = *(const v8bf*)p;
  v8bf hi = *(const v8bf*)(p + 16);
  v16bf r;
#pragma unroll
  for (int i = 0; i < 8; ++i) { r[i] = lo[i]; r[i + 8] = hi[i]; }
  return r;
}

// B-fragment (K=32 x N=16) from row-major "weight rows = output cols" source:
// lane l: row n0 + (l&15); halves at kk+off+{0..7} and kk+off+16+{0..7}, off = 8*(l>=16)
__device__ inline v16bf load_frag_B(const bf16* base, size_t stride, int n0, int kk, int lane) {
  int n = n0 + (lane & 15);
  int off = (lane < 16) ? 0 : 8;
  const bf16* p = base + (size_t)n * stride + (size_t)(kk + off);
  v8bf lo = *(const v8bf*)p;
  v8bf hi = *(const v8bf*)(p + 16);
  v16bf r;
#pragma unroll
  for (int i = 0; i < 8; ++i) { r[i] = lo[i]; r[i + 8] = hi[i]; }
  return r;
}

// ---------------- f32 -> bf16 convert ------------------------------------
__global__ void cvt_kernel(const float* __restrict__ in, bf16* __restrict__ out, int n) {
  int i = blockIdx.x * blockDim.x + threadIdx.x;
  int stride = gridDim.x * blockDim.x;
  for (; i < n; i += stride) out[i] = (bf16)in[i];
}

// ---------------- fused LayerNorm (+ optional rotary) ---------------------
__global__ __launch_bounds__(256)
void ln_rot_kernel(const float* __restrict__ x, const float* __restrict__ w,
                   const float* __restrict__ bias,
                   bf16* __restrict__ out_plain, bf16* __restrict__ out_rot) {
  __shared__ float xrow[DD];
  __shared__ float red[256];
  const int row = blockIdx.x;
  const int n = row & (NN - 1);
  const int tid = threadIdx.x;
  const float* xp = x + (size_t)row * DD;

  float v[4];
  float s = 0.f;
#pragma unroll
  for (int k = 0; k < 4; ++k) { v[k] = xp[tid + k * 256]; s += v[k]; }
  red[tid] = s; __syncthreads();
  for (int st = 128; st > 0; st >>= 1) { if (tid < st) red[tid] += red[tid + st]; __syncthreads(); }
  const float mean = red[0] * (1.f / DD);
  __syncthreads();
  float s2 = 0.f;
#pragma unroll
  for (int k = 0; k < 4; ++k) { float t = v[k] - mean; s2 += t * t; }
  red[tid] = s2; __syncthreads();
  for (int st = 128; st > 0; st >>= 1) { if (tid < st) red[tid] += red[tid + st]; __syncthreads(); }
  const float rstd = rsqrtf(red[0] * (1.f / DD) + 1e-5f);
  __syncthreads();

#pragma unroll
  for (int k = 0; k < 4; ++k) {
    int d = tid + k * 256;
    float xn = (v[k] - mean) * rstd * w[d] + bias[d];
    xrow[d] = xn;
    out_plain[(size_t)row * DD + d] = (bf16)xn;
  }
  if (out_rot != nullptr) {
    __syncthreads();
#pragma unroll
    for (int k = 0; k < 4; ++k) {
      int d = tid + k * 256;
      int j = d & 63;
      int jj = j & 31;
      int base = d - j;
      float f = (float)n * __powf(10000.f, -(float)jj * (1.f / 32.f));
      float x1 = xrow[base + jj];
      float x2 = xrow[base + 32 + jj];
      float val = (j < 32) ? (x1 - x2) * f : (x1 + x2) * f;
      out_rot[(size_t)row * DD + d] = (bf16)val;
    }
  }
}

// ---------------- generic WMMA GEMM with async-LDS double buffering -------
// Y[M,N] = A[M,K] @ Bw[N,K]^T, block tile 128x128, K-step 32.
// MODE 0: QKV projection scatter; MODE 1: resid + acc + bias -> f32 (ld 1024)
// MODE 2: gelu(acc + bias) -> bf16 (ld 4096)
#define TAPITCH 40   // padded row pitch (halves), 80B: 16B aligned

__device__ inline void gemm_prefetch(const bf16* __restrict__ Asrc,
                                     const bf16* __restrict__ Bsrc, int Kdim,
                                     int bm, int bn, int kk,
                                     bf16 (*tA)[TAPITCH], bf16 (*tB)[TAPITCH], int tid) {
  for (int c = tid; c < 512; c += 256) {
    int row = c >> 2, ch = (c & 3) * 8;
    async_copy16(Asrc + (size_t)(bm + row) * Kdim + kk + ch, &tA[row][ch]);
    async_copy16(Bsrc + (size_t)(bn + row) * Kdim + kk + ch, &tB[row][ch]);
  }
}

template <int MODE>
__global__ __launch_bounds__(256)
void gemm_kernel(const bf16* __restrict__ A, const bf16* __restrict__ A2,
                 const bf16* __restrict__ Bw, const float* __restrict__ bias,
                 const float* __restrict__ resid, float* __restrict__ outF,
                 bf16* __restrict__ outB, bf16* __restrict__ outQ,
                 bf16* __restrict__ outK, bf16* __restrict__ outVt, int Kdim) {
  __shared__ __align__(16) bf16 tA[2][128][TAPITCH];
  __shared__ __align__(16) bf16 tB[2][128][TAPITCH];
  const int tid = threadIdx.x;
  const int lane = tid & 31;
  const int wid = tid >> 5;
  const int bm = blockIdx.y * 128;
  const int bn = blockIdx.x * 128;
  const int lwm = (wid >> 2) * 64;   // wave row base within tile
  const int lwn = (wid & 3) * 32;    // wave col base within tile
  const int wm = bm + lwm;
  const int wn = bn + lwn;

  const bf16* Asrc = (MODE == 0 && bn >= 2 * DD) ? A2 : A;

  v8f acc[4][2];
#pragma unroll
  for (int i = 0; i < 4; ++i)
#pragma unroll
    for (int j = 0; j < 2; ++j)
#pragma unroll
      for (int r = 0; r < 8; ++r) acc[i][j][r] = 0.f;

  const int nk = Kdim >> 5;
  gemm_prefetch(Asrc, Bw, Kdim, bm, bn, 0, tA[0], tB[0], tid);

  for (int it = 0; it < nk; ++it) {
    wait_async0();
    __syncthreads();
    const int buf = it & 1;
    if (it + 1 < nk)
      gemm_prefetch(Asrc, Bw, Kdim, bm, bn, (it + 1) << 5, tA[buf ^ 1], tB[buf ^ 1], tid);

    v16bf af[4];
#pragma unroll
    for (int i = 0; i < 4; ++i) af[i] = load_frag_A(&tA[buf][0][0], TAPITCH, lwm + i * 16, 0, lane);
    v16bf bfr[2];
#pragma unroll
    for (int j = 0; j < 2; ++j) bfr[j] = load_frag_B(&tB[buf][0][0], TAPITCH, lwn + j * 16, 0, lane);
#pragma unroll
    for (int i = 0; i < 4; ++i)
#pragma unroll
      for (int j = 0; j < 2; ++j) acc[i][j] = wmma_bf16(af[i], bfr[j], acc[i][j]);
    __syncthreads();
  }

  const int rowoff = (lane >= 16) ? 8 : 0;
  const int col = lane & 15;
#pragma unroll
  for (int i = 0; i < 4; ++i) {
#pragma unroll
    for (int j = 0; j < 2; ++j) {
#pragma unroll
      for (int r = 0; r < 8; ++r) {
        int m = wm + i * 16 + r + rowoff;
        int nidx = wn + j * 16 + col;
        float val = acc[i][j][r] + bias[nidx];
        if (MODE == 0) {
          int b = m >> 10;
          int n = m & (NN - 1);
          if (nidx < DD) {
            int h = nidx >> 6, hd = nidx & 63;
            outQ[((size_t)(b * HH + h) * NN + n) * HDD + hd] = (bf16)val;
          } else if (nidx < 2 * DD) {
            int c = nidx - DD;
            int h = c >> 6, hd = c & 63;
            outK[((size_t)(b * HH + h) * NN + n) * HDD + hd] = (bf16)val;
          } else {
            int c = nidx - 2 * DD;
            int h = c >> 6, hd = c & 63;
            outVt[((size_t)(b * HH + h) * HDD + hd) * NN + n] = (bf16)val;  // V transposed
          }
        } else if (MODE == 1) {
          size_t idx = (size_t)m * DD + nidx;
          outF[idx] = resid[idx] + val;
        } else {
          float g = 0.5f * val * (1.f + erff(val * 0.70710678118654752f));
          outB[(size_t)m * MLPD + nidx] = (bf16)g;
        }
      }
    }
  }
}

// ---------------- flash attention with async-LDS K/V staging --------------
#define KPITCH 72   // 32x64 K tile, padded (144B rows, 16B aligned)
#define VPITCH 40   // 64x32 Vt tile, padded (80B rows)

__device__ inline void flash_prefetch(const bf16* __restrict__ Kp,
                                      const bf16* __restrict__ Vp, int j0,
                                      bf16 (*tK)[KPITCH], bf16 (*tV)[VPITCH], int tid) {
  { // K tile: 32 key rows x 64 halves = 256 x 16B chunks
    int row = tid >> 3, ch = (tid & 7) * 8;
    async_copy16(Kp + (size_t)(j0 + row) * HDD + ch, &tK[row][ch]);
  }
  { // Vt tile: 64 hd rows x 32 halves = 256 x 16B chunks
    int row = tid >> 2, ch = (tid & 3) * 8;
    async_copy16(Vp + (size_t)row * NN + j0 + ch, &tV[row][ch]);
  }
}

__global__ __launch_bounds__(256)
void flash_kernel(const bf16* __restrict__ Qb, const bf16* __restrict__ Kb,
                  const bf16* __restrict__ Vt, bf16* __restrict__ Ob) {
  __shared__ __align__(16) bf16 ldsP[8][16][32];
  __shared__ __align__(16) bf16 tK[2][32][KPITCH];
  __shared__ __align__(16) bf16 tV[2][64][VPITCH];
  const int tid = threadIdx.x;
  const int lane = tid & 31;
  const int wid = tid >> 5;
  const int bh = blockIdx.x;
  const int b = bh >> 4;
  const int h = bh & 15;
  const int q0 = blockIdx.y * 128 + wid * 16;

  const bf16* Qp = Qb + (size_t)bh * NN * HDD;
  const bf16* Kp = Kb + (size_t)bh * NN * HDD;
  const bf16* Vp = Vt + (size_t)bh * HDD * NN;

  const v16bf qa0 = load_frag_A(Qp, HDD, q0, 0, lane);
  const v16bf qa1 = load_frag_A(Qp, HDD, q0, 32, lane);

  v8f o[4];
#pragma unroll
  for (int t = 0; t < 4; ++t)
#pragma unroll
    for (int r = 0; r < 8; ++r) o[t][r] = 0.f;
  float mrun[8], lrun[8];
#pragma unroll
  for (int r = 0; r < 8; ++r) { mrun[r] = -1e30f; lrun[r] = 0.f; }

  const int rowoff = (lane >= 16) ? 8 : 0;
  const int col = lane & 15;
  const bf16* lp = &ldsP[wid][0][0];

  flash_prefetch(Kp, Vp, 0, tK[0], tV[0], tid);

  for (int it = 0; it < NN / 32; ++it) {
    const int j0 = it << 5;
    wait_async0();
    __syncthreads();
    const int buf = it & 1;
    if (it + 1 < NN / 32)
      flash_prefetch(Kp, Vp, j0 + 32, tK[buf ^ 1], tV[buf ^ 1], tid);

    const bf16* kt = &tK[buf][0][0];
    const bf16* vt = &tV[buf][0][0];
    v8f z{};
    v8f s0 = wmma_bf16(qa0, load_frag_B(kt, KPITCH, 0, 0, lane), z);
    s0 = wmma_bf16(qa1, load_frag_B(kt, KPITCH, 0, 32, lane), s0);
    v8f s1 = wmma_bf16(qa0, load_frag_B(kt, KPITCH, 16, 0, lane), z);
    s1 = wmma_bf16(qa1, load_frag_B(kt, KPITCH, 16, 32, lane), s1);

    float a0[8], a1[8], alpha[8];
#pragma unroll
    for (int r = 0; r < 8; ++r) {
      a0[r] = s0[r] * 0.125f;  // 1/sqrt(64)
      a1[r] = s1[r] * 0.125f;
      float mt = red16_max(fmaxf(a0[r], a1[r]));
      float mn = fmaxf(mrun[r], mt);
      alpha[r] = __expf(mrun[r] - mn);
      a0[r] = __expf(a0[r] - mn);
      a1[r] = __expf(a1[r] - mn);
      float rs = red16_sum(a0[r] + a1[r]);
      lrun[r] = lrun[r] * alpha[r] + rs;
      mrun[r] = mn;
    }
#pragma unroll
    for (int t = 0; t < 4; ++t)
#pragma unroll
      for (int r = 0; r < 8; ++r) o[t][r] *= alpha[r];

    // transpose P from C-layout to A-layout through LDS
#pragma unroll
    for (int r = 0; r < 8; ++r) {
      ldsP[wid][r + rowoff][col] = (bf16)a0[r];
      ldsP[wid][r + rowoff][16 + col] = (bf16)a1[r];
    }
    __syncthreads();
    v16bf pa = load_frag_A(lp, 32, 0, 0, lane);

#pragma unroll
    for (int t = 0; t < 4; ++t)
      o[t] = wmma_bf16(pa, load_frag_B(vt, VPITCH, t * 16, 0, lane), o[t]);
    __syncthreads();
  }

  float inv[8];
#pragma unroll
  for (int r = 0; r < 8; ++r) inv[r] = 1.f / lrun[r];
#pragma unroll
  for (int t = 0; t < 4; ++t) {
#pragma unroll
    for (int r = 0; r < 8; ++r) {
      int m = q0 + r + rowoff;
      int hd = t * 16 + col;
      Ob[((size_t)b * NN + m) * DD + h * HDD + hd] = (bf16)(o[t][r] * inv[r]);
    }
  }
}

// ---------------- host-side orchestration --------------------------------
extern "C" void kernel_launch(void* const* d_in, const int* in_sizes, int n_in,
                              void* d_out, int out_size, void* d_ws, size_t ws_size,
                              hipStream_t stream) {
  (void)in_sizes; (void)n_in; (void)out_size; (void)ws_size;
  const float* x        = (const float*)d_in[0];
  const float* ln1_w    = (const float*)d_in[1];
  const float* ln1_b    = (const float*)d_in[2];
  const float* in_w     = (const float*)d_in[3];
  const float* in_b     = (const float*)d_in[4];
  const float* out_w    = (const float*)d_in[5];
  const float* out_b    = (const float*)d_in[6];
  const float* ln2_w    = (const float*)d_in[7];
  const float* ln2_b    = (const float*)d_in[8];
  const float* mlp_w1   = (const float*)d_in[9];
  const float* mlp_b1   = (const float*)d_in[10];
  const float* mlp_w2   = (const float*)d_in[11];
  const float* mlp_b2   = (const float*)d_in[12];

  char* ws = (char*)d_ws;
  size_t off = 0;
  auto alloc = [&](size_t bytes) { char* p = ws + off; off += (bytes + 255) & ~(size_t)255; return p; };
  bf16* wQK   = (bf16*)alloc((size_t)3 * DD * DD * 2);
  bf16* wO    = (bf16*)alloc((size_t)DD * DD * 2);
  bf16* wM1   = (bf16*)alloc((size_t)MLPD * DD * 2);
  bf16* wM2   = (bf16*)alloc((size_t)DD * MLPD * 2);
  bf16* qk_in = (bf16*)alloc((size_t)ROWS * DD * 2);
  bf16* v_in  = (bf16*)alloc((size_t)ROWS * DD * 2);
  bf16* Qb    = (bf16*)alloc((size_t)ROWS * DD * 2);
  bf16* Kb    = (bf16*)alloc((size_t)ROWS * DD * 2);
  bf16* Vt    = (bf16*)alloc((size_t)ROWS * DD * 2);
  bf16* Ob    = (bf16*)alloc((size_t)ROWS * DD * 2);
  float* x1   = (float*)alloc((size_t)ROWS * DD * 4);
  bf16* xn2   = (bf16*)alloc((size_t)ROWS * DD * 2);
  bf16* hmid  = (bf16*)alloc((size_t)ROWS * MLPD * 2);

  cvt_kernel<<<2048, 256, 0, stream>>>(in_w,  wQK, 3 * DD * DD);
  cvt_kernel<<<1024, 256, 0, stream>>>(out_w, wO,  DD * DD);
  cvt_kernel<<<2048, 256, 0, stream>>>(mlp_w1, wM1, MLPD * DD);
  cvt_kernel<<<2048, 256, 0, stream>>>(mlp_w2, wM2, DD * MLPD);

  ln_rot_kernel<<<ROWS, 256, 0, stream>>>(x, ln1_w, ln1_b, v_in, qk_in);

  gemm_kernel<0><<<dim3(3 * DD / 128, ROWS / 128), 256, 0, stream>>>(
      qk_in, v_in, wQK, in_b, nullptr, nullptr, nullptr, Qb, Kb, Vt, DD);

  flash_kernel<<<dim3(BB * HH, NN / 128), 256, 0, stream>>>(Qb, Kb, Vt, Ob);

  gemm_kernel<1><<<dim3(DD / 128, ROWS / 128), 256, 0, stream>>>(
      Ob, nullptr, wO, out_b, x, x1, nullptr, nullptr, nullptr, nullptr, DD);

  ln_rot_kernel<<<ROWS, 256, 0, stream>>>(x1, ln2_w, ln2_b, xn2, nullptr);

  gemm_kernel<2><<<dim3(MLPD / 128, ROWS / 128), 256, 0, stream>>>(
      xn2, nullptr, wM1, mlp_b1, nullptr, nullptr, hmid, nullptr, nullptr, nullptr, DD);

  gemm_kernel<1><<<dim3(DD / 128, ROWS / 128), 256, 0, stream>>>(
      hmid, nullptr, wM2, mlp_b2, x1, (float*)d_out, nullptr, nullptr, nullptr, nullptr, MLPD);
}
